// VectorQuantize_44856638440089
// MI455X (gfx1250) — matrix operand order repaired and checked by reading, not compiled, not run on hardware
//
#include <hip/hip_runtime.h>

// Problem constants (from reference): H=1, N=32768, D=256, K=4096, all fp32.
#define NN 32768
#define DD 256
#define KK 4096

typedef __attribute__((ext_vector_type(16))) __bf16          v16bf;
typedef __attribute__((ext_vector_type(16))) unsigned short  v16u;
typedef __attribute__((ext_vector_type(8)))  float           v8f;

static __device__ __forceinline__ unsigned short f32_to_bf16_rne(float f) {
    unsigned u = __float_as_uint(f);
    u += 0x7FFFu + ((u >> 16) & 1u);        // round-to-nearest-even
    return (unsigned short)(u >> 16);
}
static __device__ __forceinline__ float bf16_to_f32(unsigned short h) {
    return __uint_as_float(((unsigned)h) << 16);
}
static __device__ __forceinline__ v16bf as_bf16x16(v16u u) {
    union { v16u u; v16bf b; } c; c.u = u; return c.b;
}

// ---------------------------------------------------------------------------
// Prep: convert embed rows into WMMA B-fragment layout (bf16 hi + lo split).
// B (32x16, 16-bit, wave32): lane l holds column n=l&15; elements e=0..15 map
// to K = dcBase + e + ((l&16)?16:0)  (contiguous 16 K values per lane).
// Workspace layout: [K/16 chunks][sel(hi,lo)][dc 0..7][lane 0..31][16 bf16]
//   => 8192 bf16 = 16 KiB per code chunk, 4 MiB total.
// One block (256 thr) per 16-code chunk: thread = (dc, lane).
// ---------------------------------------------------------------------------
__global__ void vq_prep_embed(const float* __restrict__ embed,
                              unsigned short* __restrict__ wsB) {
    const int kc   = blockIdx.x;          // 0..255
    const int dc   = threadIdx.x >> 5;    // 0..7
    const int lane = threadIdx.x & 31;
    const int code = kc * 16 + (lane & 15);
    const int kb   = dc * 32 + ((lane & 16) ? 16 : 0);

    const float4* src = (const float4*)(embed + (size_t)code * DD + kb);
    float f[16];
#pragma unroll
    for (int i = 0; i < 4; ++i) {
        float4 v = src[i];
        f[i*4+0] = v.x; f[i*4+1] = v.y; f[i*4+2] = v.z; f[i*4+3] = v.w;
    }
    v16u hi, lo;
#pragma unroll
    for (int e = 0; e < 16; ++e) {
        unsigned short h = f32_to_bf16_rne(f[e]);
        unsigned short l = f32_to_bf16_rne(f[e] - bf16_to_f32(h));
        hi[e] = h; lo[e] = l;
    }
    unsigned short* chunk = wsB + (size_t)kc * 8192;
    *(v16u*)(chunk + ((size_t)(0*8 + dc) * 32 + lane) * 16) = hi;
    *(v16u*)(chunk + ((size_t)(1*8 + dc) * 32 + lane) * 16) = lo;
}

// e2[k] = sum_d embed[k][d]^2  (exact fp32)
__global__ void vq_e2(const float* __restrict__ embed, float* __restrict__ e2) {
    const int k = blockIdx.x * blockDim.x + threadIdx.x;
    const float4* row = (const float4*)(embed + (size_t)k * DD);
    float s = 0.f;
#pragma unroll 4
    for (int i = 0; i < DD / 4; ++i) {
        float4 v = row[i];
        s += v.x*v.x + v.y*v.y + v.z*v.z + v.w*v.w;
    }
    e2[k] = s;
}

// ---------------------------------------------------------------------------
// Main: per-wave 16 tokens, sweep all 4096 codes in chunks of 16.
// xe computed via bf16x3 WMMA (hi*hi + hi*lo + lo*hi), fp32 accumulate.
// B fragments staged through LDS (16 KiB / chunk) and shared by all 8 waves.
// ---------------------------------------------------------------------------
static __device__ __forceinline__ v16bf load_frag_lds(const uint4* ldsB,
                                                      int sel, int dc, int lane) {
    const uint4* p = ldsB + (((sel*8 + dc) * 32 + lane) << 1); // 32B per frag
    union { uint4 q[2]; v16u u; } c;
    c.q[0] = p[0]; c.q[1] = p[1];
    return as_bf16x16(c.u);
}

__global__ void __launch_bounds__(256, 1) vq_argmin(
    const float* __restrict__ x,
    const unsigned short* __restrict__ wsB,
    const float* __restrict__ e2,
    unsigned int* __restrict__ idxOut)
{
    __shared__ uint4 ldsB[1024];          // 16 KiB: one code-chunk of B frags
    const int tid    = threadIdx.x;
    const int wave   = tid >> 5;
    const int lane   = tid & 31;
    const int m      = lane & 15;
    const int hiHalf = (lane >> 4) & 1;
    const int n0     = blockIdx.x * 128 + wave * 16;   // this wave's 16 tokens

    // Preload A fragments (16x32 bf16 per d-chunk). A layout: lane holds row
    // m=l&15; elements e -> K = e + (hiHalf?16:0).
    const float* xrow = x + (size_t)(n0 + m) * DD;
    v16bf ahi[8], alo[8];
#pragma unroll
    for (int dc = 0; dc < 8; ++dc) {
        const int kb = dc * 32 + (hiHalf ? 16 : 0);
        float f[16];
        {
            float4 a = *(const float4*)(xrow + kb +  0);
            float4 b = *(const float4*)(xrow + kb +  4);
            float4 c = *(const float4*)(xrow + kb +  8);
            float4 d = *(const float4*)(xrow + kb + 12);
            f[0]=a.x; f[1]=a.y; f[2]=a.z; f[3]=a.w;
            f[4]=b.x; f[5]=b.y; f[6]=b.z; f[7]=b.w;
            f[8]=c.x; f[9]=c.y; f[10]=c.z; f[11]=c.w;
            f[12]=d.x; f[13]=d.y; f[14]=d.z; f[15]=d.w;
        }
        v16u h, l;
#pragma unroll
        for (int e = 0; e < 16; ++e) {
            unsigned short hh = f32_to_bf16_rne(f[e]);
            unsigned short ll = f32_to_bf16_rne(f[e] - bf16_to_f32(hh));
            h[e] = hh; l[e] = ll;
        }
        ahi[dc] = as_bf16x16(h);
        alo[dc] = as_bf16x16(l);
    }

    float    bv[8];
    unsigned bi[8];
#pragma unroll
    for (int r = 0; r < 8; ++r) { bv[r] = 3.4e38f; bi[r] = 0u; }

    const uint4* bsrc = (const uint4*)wsB;
    for (int kc = 0; kc < KK / 16; ++kc) {
        __syncthreads();
        const uint4* src = bsrc + (size_t)kc * 1024 + tid * 4;
#pragma unroll
        for (int i = 0; i < 4; ++i) ldsB[tid * 4 + i] = src[i];
        __syncthreads();

        v8f acc = {};
#pragma unroll
        for (int dc = 0; dc < 8; ++dc) {
            v16bf bh = load_frag_lds(ldsB, 0, dc, lane);
            v16bf bl = load_frag_lds(ldsB, 1, dc, lane);
            acc = __builtin_amdgcn_wmma_f32_16x16x32_bf16(
                      false, alo[dc], false, bh, (short)0, acc, false, false);
            acc = __builtin_amdgcn_wmma_f32_16x16x32_bf16(
                      false, ahi[dc], false, bl, (short)0, acc, false, false);
            acc = __builtin_amdgcn_wmma_f32_16x16x32_bf16(
                      false, ahi[dc], false, bh, (short)0, acc, false, false);
        }

        const float    e2v  = e2[kc * 16 + m];
        const unsigned code = (unsigned)(kc * 16 + m);
#pragma unroll
        for (int r = 0; r < 8; ++r) {
            float score = __builtin_fmaf(-2.f, acc[r], e2v); // e2 - 2*xe
            if (score < bv[r]) { bv[r] = score; bi[r] = code; }
        }
    }

    // Reduce across the 16 lanes of each half (xor butterfly stays in-half).
#pragma unroll
    for (int r = 0; r < 8; ++r) {
#pragma unroll
        for (int off = 1; off < 16; off <<= 1) {
            float    ov = __shfl_xor(bv[r], off, 32);
            unsigned oi = (unsigned)__shfl_xor((int)bi[r], off, 32);
            if (ov < bv[r] || (ov == bv[r] && oi < bi[r])) { bv[r] = ov; bi[r] = oi; }
        }
        if (m == 0) idxOut[n0 + r + hiHalf * 8] = bi[r];
    }
}

// Gather: out[n][:] = embed[idx[n]][:]  (fp32, float4 copies)
__global__ void vq_gather(const float* __restrict__ embed,
                          const unsigned int* __restrict__ idx,
                          float* __restrict__ out) {
    const int token = blockIdx.x * 4 + (threadIdx.x >> 6);
    const int d     = (threadIdx.x & 63) * 4;
    const unsigned c = idx[token];
    *(float4*)(out + (size_t)token * DD + d) =
        *(const float4*)(embed + (size_t)c * DD + d);
}

extern "C" void kernel_launch(void* const* d_in, const int* in_sizes, int n_in,
                              void* d_out, int out_size, void* d_ws, size_t ws_size,
                              hipStream_t stream) {
    const float* x     = (const float*)d_in[0];   // [1, 32768, 256]
    const float* embed = (const float*)d_in[1];   // [1, 4096, 256]
    char* ws = (char*)d_ws;
    unsigned short* wsB = (unsigned short*)ws;                        // 4 MiB frags
    float*        e2p   = (float*)(ws + (size_t)4 * 1024 * 1024);     // 16 KiB
    unsigned int* idx   = (unsigned int*)(ws + (size_t)4 * 1024 * 1024 + 16384);
    float* out = (float*)d_out;

    vq_prep_embed<<<KK / 16,  256, 0, stream>>>(embed, wsB);
    vq_e2        <<<KK / 256, 256, 0, stream>>>(embed, e2p);
    vq_argmin    <<<NN / 128, 256, 0, stream>>>(x, wsB, e2p, idx);
    vq_gather    <<<NN / 4,   256, 0, stream>>>(embed, idx, out);
}
